// CrossAttention3D_4552665334068
// MI455X (gfx1250) — matrix operand order repaired and checked by reading, not compiled, not run on hardware
//
#include <hip/hip_runtime.h>
#include <hip/hip_bf16.h>

typedef __bf16 bf16;
typedef __attribute__((ext_vector_type(16))) __bf16 v16bf;
typedef __attribute__((ext_vector_type(8)))  __bf16 v8bf;
typedef __attribute__((ext_vector_type(8)))  float  v8f;

#define BB  2
#define QQ  2048
#define KVN 2048
#define DD  1024
#define HH  16
#define DHD 64

// ---------------- WMMA fragment helpers (CDNA5 16x16x32 bf16 layouts) -------
// A (16x32, 16-bit): lane row = lane&15; 16 contiguous K at (lane>>4)*16.
static __device__ __forceinline__ v16bf load_a_frag(const bf16* src, int ld, int k0, int lane) {
  int row = lane & 15;
  int g   = lane >> 4;
  const bf16* p = src + (size_t)row * ld + k0 + g * 16;
  return *(const v16bf*)(p);
}

// B (32x16, 16-bit): lane column n = lane&15; 16 contiguous K at (lane>>4)*16.
// src is row-major [N][K] (i.e. B transposed in memory == torch Linear weight / K rows / V^T).
static __device__ __forceinline__ v16bf load_b_frag(const bf16* src, int n_base, int ld, int k0, int lane) {
  int n  = n_base + (lane & 15);
  int kb = k0 + (lane >> 4) * 16;
  return *(const v16bf*)(src + (size_t)n * ld + kb);
}

static __device__ __forceinline__ v8f wmma_bf16(v16bf a, v16bf b, v8f c) {
  return __builtin_amdgcn_wmma_f32_16x16x32_bf16(false, a, false, b, (short)0, c, false, false);
}

// ---------------- CDNA5 async global->LDS (ASYNCcnt path) --------------------
// Flat LDS addresses carry the LDS offset in addr[31:0] (ISA 10.2), so the
// low 32 bits of a generic pointer to __shared__ are a valid LDS address.
static __device__ __forceinline__ void async_copy_b128(void* lds_dst, const void* gsrc) {
  unsigned int  ldsa = (unsigned int)(unsigned long long)(uintptr_t)lds_dst;
  unsigned long long ga = (unsigned long long)(uintptr_t)gsrc;
  asm volatile("global_load_async_to_lds_b128 %0, %1, off"
               :: "v"(ldsa), "v"(ga) : "memory");
}
static __device__ __forceinline__ void wait_asynccnt0() {
#if __has_builtin(__builtin_amdgcn_s_wait_asynccnt)
  __builtin_amdgcn_s_wait_asynccnt(0);
#else
  asm volatile("s_wait_asynccnt 0x0" ::: "memory");
#endif
}

// ---------------- elementwise convert (weights) ------------------------------
__global__ void k_convert_bf16(const float* __restrict__ in, bf16* __restrict__ out, int n) {
  int i = blockIdx.x * blockDim.x + threadIdx.x;
  if (i < n) out[i] = (bf16)in[i];
}

// ---------------- RMSNorm rows of D=1024 -> bf16 -----------------------------
__global__ void k_rmsnorm_bf16(const float* __restrict__ x, const float* __restrict__ w,
                               bf16* __restrict__ out, float eps) {
  __shared__ float red[256];
  int row = blockIdx.x;
  const float* xr = x + (size_t)row * DD;
  float v[4];
  float ss = 0.f;
  #pragma unroll
  for (int i = 0; i < 4; ++i) { v[i] = xr[threadIdx.x + i * 256]; ss += v[i] * v[i]; }
  red[threadIdx.x] = ss;
  __syncthreads();
  for (int off = 128; off > 0; off >>= 1) {
    if ((int)threadIdx.x < off) red[threadIdx.x] += red[threadIdx.x + off];
    __syncthreads();
  }
  float r = rsqrtf(red[0] * (1.0f / (float)DD) + eps);
  bf16* orow = out + (size_t)row * DD;
  #pragma unroll
  for (int i = 0; i < 4; ++i) {
    int c = threadIdx.x + i * 256;
    orow[c] = (bf16)(v[i] * r * w[c]);
  }
}

// ---------------- V [B,KV,D] f32 -> V^T [B,D,KV] bf16 ------------------------
__global__ void k_transpose_v(const float* __restrict__ v, bf16* __restrict__ vt) {
  __shared__ float tile[32][33];
  int b  = blockIdx.z;
  int d0 = blockIdx.x * 32;
  int k0 = blockIdx.y * 32;
  int tx = threadIdx.x, ty = threadIdx.y;   // 32 x 8
  const float* vb = v + (size_t)b * KVN * DD;
  bf16* vtb = vt + (size_t)b * DD * KVN;
  #pragma unroll
  for (int j = 0; j < 32; j += 8)
    tile[ty + j][tx] = vb[(size_t)(k0 + ty + j) * DD + d0 + tx];
  __syncthreads();
  #pragma unroll
  for (int j = 0; j < 32; j += 8)
    vtb[(size_t)(d0 + ty + j) * KVN + k0 + tx] = (bf16)tile[tx][ty + j];
}

// ---------------- projection GEMM: C = scale*(A @ W^T + bias) (bf16 out) -----
// Ping-pong software pipeline with peeled prologue/epilogue: loads write the
// loop-carried fragment registers directly (no phi-rotation movs, no WAR nops).
__global__ void __launch_bounds__(256, 2) k_proj_gemm(
    const bf16* __restrict__ A, const bf16* __restrict__ W,
    const float* __restrict__ bias, bf16* __restrict__ C, float scale) {
  int lane = threadIdx.x & 31;
  int wv   = threadIdx.x >> 5;                    // 8 waves: 2 (m) x 4 (n)
  int m0 = blockIdx.x * 32 + (wv >> 2) * 16;
  int n0 = blockIdx.y * 256 + (wv & 3) * 64;
  v8f acc[4] = {};
  const bf16* Arow = A + (size_t)m0 * DD;

  v16bf a0 = load_a_frag(Arow, DD, 0, lane);
  v16bf a1 = load_a_frag(Arow, DD, 32, lane);
  v16bf b0[4], b1[4];
  #pragma unroll
  for (int nt = 0; nt < 4; ++nt) {
    b0[nt] = load_b_frag(W, n0 + nt * 16, DD, 0,  lane);
    b1[nt] = load_b_frag(W, n0 + nt * 16, DD, 32, lane);
  }

  for (int kk = 0; kk + 64 < DD; kk += 64) {
    __builtin_prefetch(Arow + kk + 128, 0, 3);
    #pragma unroll
    for (int nt = 0; nt < 4; ++nt) acc[nt] = wmma_bf16(a0, b0[nt], acc[nt]);
    a0 = load_a_frag(Arow, DD, kk + 64, lane);
    #pragma unroll
    for (int nt = 0; nt < 4; ++nt) b0[nt] = load_b_frag(W, n0 + nt * 16, DD, kk + 64, lane);
    #pragma unroll
    for (int nt = 0; nt < 4; ++nt) acc[nt] = wmma_bf16(a1, b1[nt], acc[nt]);
    a1 = load_a_frag(Arow, DD, kk + 96, lane);
    #pragma unroll
    for (int nt = 0; nt < 4; ++nt) b1[nt] = load_b_frag(W, n0 + nt * 16, DD, kk + 96, lane);
  }
  #pragma unroll
  for (int nt = 0; nt < 4; ++nt) acc[nt] = wmma_bf16(a0, b0[nt], acc[nt]);
  #pragma unroll
  for (int nt = 0; nt < 4; ++nt) acc[nt] = wmma_bf16(a1, b1[nt], acc[nt]);

  int g = lane >> 4;
  #pragma unroll
  for (int nt = 0; nt < 4; ++nt) {
    int n = n0 + nt * 16 + (lane & 15);
    float bn = bias[n];
    #pragma unroll
    for (int v = 0; v < 8; ++v) {
      int m = m0 + v + 8 * g;
      C[(size_t)m * DD + n] = (bf16)((acc[nt][v] + bn) * scale);
    }
  }
}

// ---------------- fused attention ------------------------------------------
// One 16-row q tile per workgroup; 16 waves. Per head, each wave computes the
// scores of exactly the 128 keys whose P entries it owns in pass 2
// (kv = blk*512 + wave*32 + t2*16), keeps them in registers, so scores are
// computed exactly once. P is pre-scaled by 1/(sum*H) (head-average folded).
__global__ void __launch_bounds__(512, 1) k_attention(
    const bf16* __restrict__ qp, const bf16* __restrict__ kp,
    const bf16* __restrict__ vt, float* __restrict__ out) {
  __shared__ bf16  q_lds[16 * DD];      // 32 KB: q tile, all heads
  __shared__ bf16  p_lds[16 * 512];     // 16 KB: P block
  __shared__ float red_max[16 * 16];
  __shared__ float red_sum[16 * 16];

  int lane = threadIdx.x & 31;
  int wv   = threadIdx.x >> 5;          // 16 waves
  int g    = lane >> 4;
  int b    = blockIdx.y;
  int q0   = blockIdx.x * 16;
  int d0   = wv * 64;                   // this wave's output d-slice

  const bf16* qpb = qp + ((size_t)b * QQ + q0) * DD;
  const bf16* kpb = kp + (size_t)b * KVN * DD;
  const bf16* vtb = vt + (size_t)b * DD * KVN;

  // async global->LDS staging of the q tile: 4 x b128 per thread (ASYNCcnt)
  for (int i = threadIdx.x; i < 16 * DD / 8; i += 512)
    async_copy_b128(&q_lds[i * 8], qpb + (size_t)i * 8);
  wait_asynccnt0();
  __syncthreads();

  v8f o[4] = {};

  for (int h = 0; h < HH; ++h) {
    int hb = h * DHD;
    v16bf af0 = load_a_frag(q_lds, DD, hb, lane);
    v16bf af1 = load_a_frag(q_lds, DD, hb + 32, lane);

    // ---- pass 1: scores for this wave's 8 tiles (kept in registers) ----
    v8f s[8];
    #pragma unroll
    for (int t = 0; t < 8; ++t) {
      int kv0 = (t >> 1) * 512 + wv * 32 + (t & 1) * 16;
      v8f acc = {};
      acc = wmma_bf16(af0, load_b_frag(kpb, kv0, DD, hb,      lane), acc);
      acc = wmma_bf16(af1, load_b_frag(kpb, kv0, DD, hb + 32, lane), acc);
      s[t] = acc;
    }

    // wave-local row max over 8 tiles + 16 lanes
    float rmax[8], rsum[8];
    #pragma unroll
    for (int v = 0; v < 8; ++v) {
      float m = s[0][v];
      #pragma unroll
      for (int t = 1; t < 8; ++t) m = fmaxf(m, s[t][v]);
      m = fmaxf(m, __shfl_xor(m, 1, 16));
      m = fmaxf(m, __shfl_xor(m, 2, 16));
      m = fmaxf(m, __shfl_xor(m, 4, 16));
      m = fmaxf(m, __shfl_xor(m, 8, 16));
      rmax[v] = m;
    }
    // exp in place + wave-local row sum
    #pragma unroll
    for (int v = 0; v < 8; ++v) {
      float sum = 0.f;
      #pragma unroll
      for (int t = 0; t < 8; ++t) {
        float e = __expf(s[t][v] - rmax[v]);
        s[t][v] = e;
        sum += e;
      }
      sum += __shfl_xor(sum, 1, 16);
      sum += __shfl_xor(sum, 2, 16);
      sum += __shfl_xor(sum, 4, 16);
      sum += __shfl_xor(sum, 8, 16);
      rsum[v] = sum;
    }
    if ((lane & 15) == 0) {
      #pragma unroll
      for (int v = 0; v < 8; ++v) {
        int row = v + 8 * g;
        red_max[wv * 16 + row] = rmax[v];
        red_sum[wv * 16 + row] = rsum[v];
      }
    }
    __syncthreads();
    // combine across waves; f = exp(rmax - gmax) / (gsum * H)
    float f[8];
    #pragma unroll
    for (int v = 0; v < 8; ++v) {
      int row = v + 8 * g;
      float gm = -3.0e38f;
      for (int ww = 0; ww < 16; ++ww) gm = fmaxf(gm, red_max[ww * 16 + row]);
      float gs = 0.f;
      for (int ww = 0; ww < 16; ++ww)
        gs += red_sum[ww * 16 + row] * __expf(red_max[ww * 16 + row] - gm);
      f[v] = __expf(rmax[v] - gm) / (gs * (float)HH);
    }
    __syncthreads();

    // ---- pass 2: write P from registers per 512-KV block, then P @ V^T ----
    #pragma unroll
    for (int blk = 0; blk < 4; ++blk) {
      int kb0 = blk * 512;
      #pragma unroll
      for (int t2 = 0; t2 < 2; ++t2) {
        #pragma unroll
        for (int v = 0; v < 8; ++v) {
          p_lds[(size_t)(v + 8 * g) * 512 + wv * 32 + t2 * 16 + (lane & 15)] =
              (bf16)(s[blk * 2 + t2][v] * f[v]);
        }
      }
      __syncthreads();
      // P @ V^T, peeled ping-pong over 16 K-steps of 32
      v16bf bvA[4], bvB[4];
      #pragma unroll
      for (int nt = 0; nt < 4; ++nt) {
        bvA[nt] = load_b_frag(vtb, d0 + nt * 16, KVN, kb0,      lane);
        bvB[nt] = load_b_frag(vtb, d0 + nt * 16, KVN, kb0 + 32, lane);
      }
      for (int ks = 0; ks < 14; ks += 2) {
        v16bf pa = load_a_frag(p_lds, 512, ks * 32, lane);
        #pragma unroll
        for (int nt = 0; nt < 4; ++nt) o[nt] = wmma_bf16(pa, bvA[nt], o[nt]);
        #pragma unroll
        for (int nt = 0; nt < 4; ++nt)
          bvA[nt] = load_b_frag(vtb, d0 + nt * 16, KVN, kb0 + (ks + 2) * 32, lane);
        pa = load_a_frag(p_lds, 512, (ks + 1) * 32, lane);
        #pragma unroll
        for (int nt = 0; nt < 4; ++nt) o[nt] = wmma_bf16(pa, bvB[nt], o[nt]);
        #pragma unroll
        for (int nt = 0; nt < 4; ++nt)
          bvB[nt] = load_b_frag(vtb, d0 + nt * 16, KVN, kb0 + (ks + 3) * 32, lane);
      }
      {
        v16bf pa = load_a_frag(p_lds, 512, 14 * 32, lane);
        #pragma unroll
        for (int nt = 0; nt < 4; ++nt) o[nt] = wmma_bf16(pa, bvA[nt], o[nt]);
        pa = load_a_frag(p_lds, 512, 15 * 32, lane);
        #pragma unroll
        for (int nt = 0; nt < 4; ++nt) o[nt] = wmma_bf16(pa, bvB[nt], o[nt]);
      }
      __syncthreads();
    }
  }

  float* ob = out + ((size_t)b * QQ + q0) * DD;
  #pragma unroll
  for (int nt = 0; nt < 4; ++nt) {
    int n = d0 + nt * 16 + (lane & 15);
    #pragma unroll
    for (int v = 0; v < 8; ++v)
      ob[(size_t)(v + 8 * g) * DD + n] = o[nt][v];
  }
}

// ---------------------------------------------------------------------------
extern "C" void kernel_launch(void* const* d_in, const int* in_sizes, int n_in,
                              void* d_out, int out_size, void* d_ws, size_t ws_size,
                              hipStream_t stream) {
  const float* query = (const float*)d_in[0];
  const float* key   = (const float*)d_in[1];
  const float* value = (const float*)d_in[2];
  const float* wq_n  = (const float*)d_in[3];
  const float* wk_n  = (const float*)d_in[4];
  const float* Wq    = (const float*)d_in[5];
  const float* Wk    = (const float*)d_in[6];
  const float* bq    = (const float*)d_in[7];
  const float* bk    = (const float*)d_in[8];
  (void)in_sizes; (void)n_in; (void)out_size; (void)ws_size;

  char* ws = (char*)d_ws;
  const size_t SZ_ROWS = (size_t)BB * QQ * DD * sizeof(bf16);   // 8 MB
  const size_t SZ_W    = (size_t)DD * DD * sizeof(bf16);        // 2 MB
  bf16* qn  = (bf16*)(ws);
  bf16* kn  = (bf16*)(ws + SZ_ROWS);
  bf16* wqb = (bf16*)(ws + 2 * SZ_ROWS);
  bf16* wkb = (bf16*)(ws + 2 * SZ_ROWS + SZ_W);
  bf16* qpb = (bf16*)(ws + 2 * SZ_ROWS + 2 * SZ_W);
  bf16* kpb = (bf16*)(ws + 3 * SZ_ROWS + 2 * SZ_W);
  bf16* vtb = (bf16*)(ws + 4 * SZ_ROWS + 2 * SZ_W);
  float* out = (float*)d_out;

  const float eps = 1.1920929e-07f;   // jnp.finfo(float32).eps

  // 1) weight conversion
  k_convert_bf16<<<(DD * DD + 255) / 256, 256, 0, stream>>>(Wq, wqb, DD * DD);
  k_convert_bf16<<<(DD * DD + 255) / 256, 256, 0, stream>>>(Wk, wkb, DD * DD);

  // 2) RMSNorm q, k -> bf16
  k_rmsnorm_bf16<<<BB * QQ,  256, 0, stream>>>(query, wq_n, qn, eps);
  k_rmsnorm_bf16<<<BB * KVN, 256, 0, stream>>>(key,   wk_n, kn, eps);

  // 3) V -> V^T bf16
  {
    dim3 gr(DD / 32, KVN / 32, BB);
    dim3 bl(32, 8);
    k_transpose_v<<<gr, bl, 0, stream>>>(value, vtb);
  }

  // 4) projections (1/sqrt(DH)=0.125 folded into q)
  {
    dim3 gr(BB * QQ / 32, DD / 256);
    k_proj_gemm<<<gr, 256, 0, stream>>>(qn, wqb, bq, qpb, 0.125f);
    k_proj_gemm<<<gr, 256, 0, stream>>>(kn, wkb, bk, kpb, 1.0f);
  }

  // 5) fused attention
  {
    dim3 gr(QQ / 16, BB);
    k_attention<<<gr, 512, 0, stream>>>(qpb, kpb, vtb, out);
  }
}
